// LatentBridgeV15_65575560675407
// MI455X (gfx1250) — matrix operand order repaired and checked by reading, not compiled, not run on hardware
//
#include <hip/hip_runtime.h>
#include <hip/hip_bf16.h>

// ---------------------------------------------------------------------------
// Problem constants (match reference)
// ---------------------------------------------------------------------------
#define BB 4
#define TT 2048
#define SRCD 4096
#define DD 2048
#define KK 128
#define HH 8
#define LL 4
#define NEE 4096
#define HDD (DD / HH)   // 256
#define FFD (4 * DD)    // 8192

typedef __attribute__((ext_vector_type(16))) __bf16 bf16x16;
typedef __attribute__((ext_vector_type(8)))  __bf16 bf16x8;
typedef __attribute__((ext_vector_type(4)))  __bf16 bf16x4;
typedef __attribute__((ext_vector_type(8)))  float  v8f;

// ---------------------------------------------------------------------------
// Generic batched WMMA GEMM:  C = act(alpha * (A @ B^T) + bias) + residual
//   A indexed: A[zoff + m*sAm + k*sAk]   (all call sites use sAk == 1)
//   B indexed: B[zoff + n*sBn + k*sBk]   (sBk==1 OR sBn==1)
//   C indexed: C[zoff + m*sCm + n]
// REQUIRES: M % 128 == 0, N % 64 == 0, Kd % 64 == 0 (true for all call sites).
// Block tile 128x64, K-tile 64, 256 threads (8 waves, 4x2), wave -> 32x32.
// LDS double-buffered; next K-tile fetched into registers during WMMA.
// ---------------------------------------------------------------------------
#define BM 128
#define BN 64
#define BKT 64
#define LDSP 72   // row pitch (144 B) -> all fragment subtiles 16B aligned

__device__ __forceinline__ bf16x16 ldsfrag(const __bf16* p)
{
    bf16x8 lo = *(const bf16x8*)p;
    bf16x8 hi = *(const bf16x8*)(p + 8);
    return __builtin_shufflevector(lo, hi, 0,1,2,3,4,5,6,7,8,9,10,11,12,13,14,15);
}

__global__ __launch_bounds__(256)
void wmma_gemm_bf16(const float* __restrict__ A,
                    long long sAm, long long sAk, long long sAhi, long long sAlo,
                    const float* __restrict__ Bp,
                    long long sBn, long long sBk, long long sBhi, long long sBlo,
                    float* __restrict__ C,
                    long long sCm, long long sChi, long long sClo,
                    const float* __restrict__ bias,
                    const float* __restrict__ Res,
                    int M, int N, int Kd, float alpha, int act, int nlo)
{
    __shared__ __attribute__((aligned(16))) __bf16 As[2][BM][LDSP];
    __shared__ __attribute__((aligned(16))) __bf16 Bs[2][BN][LDSP];

    const int z  = blockIdx.z;
    const int zh = z / nlo;
    const int zl = z - zh * nlo;
    A  += (long long)zh * sAhi + (long long)zl * sAlo;
    Bp += (long long)zh * sBhi + (long long)zl * sBlo;
    const long long coff = (long long)zh * sChi + (long long)zl * sClo;
    C += coff;
    if (Res) Res += coff;

    const int m0   = blockIdx.y * BM;
    const int n0   = blockIdx.x * BN;
    const int tid  = threadIdx.x;
    const int lane = tid & 31;
    const int wave = tid >> 5;
    const int mo   = (wave & 3) << 5;   // 0,32,64,96
    const int no   = (wave >> 2) << 5;  // 0,32
    const int half = lane >> 4;         // 0/1
    const int r    = lane & 15;

    const bool bKContig = (sBk == 1);   // else sBn == 1 (attention @ V)

    // staging registers: A tile 128x64 -> 8 float4/thread; B tile 64x64 -> 4
    float4 ra[8], rb[4];

    auto fetch = [&](int k0) {
        #pragma unroll
        for (int j = 0; j < 8; ++j) {
            const int id = tid * 8 + j;          // 0..2047
            const int rr = id >> 4;              // 0..127
            const int cc = (id & 15) << 2;       // 0..60
            ra[j] = *(const float4*)&A[(long long)(m0 + rr) * sAm + (k0 + cc)];
        }
        if (bKContig) {
            #pragma unroll
            for (int j = 0; j < 4; ++j) {
                const int id = tid * 4 + j;      // 0..1023
                const int rr = id >> 4;          // n row 0..63
                const int cc = (id & 15) << 2;
                rb[j] = *(const float4*)&Bp[(long long)(n0 + rr) * sBn + (k0 + cc)];
            }
        } else {
            #pragma unroll
            for (int j = 0; j < 4; ++j) {
                const int id = tid * 4 + j;
                const int kk = id >> 4;          // k row 0..63
                const int nn = (id & 15) << 2;   // contiguous n
                rb[j] = *(const float4*)&Bp[(long long)(k0 + kk) * sBk + (n0 + nn)];
            }
        }
    };

    auto stage = [&](int buf) {
        #pragma unroll
        for (int j = 0; j < 8; ++j) {
            const int id = tid * 8 + j;
            const int rr = id >> 4;
            const int cc = (id & 15) << 2;
            bf16x4 c;
            c[0] = (__bf16)ra[j].x; c[1] = (__bf16)ra[j].y;
            c[2] = (__bf16)ra[j].z; c[3] = (__bf16)ra[j].w;
            *(bf16x4*)&As[buf][rr][cc] = c;
        }
        if (bKContig) {
            #pragma unroll
            for (int j = 0; j < 4; ++j) {
                const int id = tid * 4 + j;
                const int rr = id >> 4;
                const int cc = (id & 15) << 2;
                bf16x4 c;
                c[0] = (__bf16)rb[j].x; c[1] = (__bf16)rb[j].y;
                c[2] = (__bf16)rb[j].z; c[3] = (__bf16)rb[j].w;
                *(bf16x4*)&Bs[buf][rr][cc] = c;
            }
        } else {
            // transpose-scatter: Bs[n][k]
            #pragma unroll
            for (int j = 0; j < 4; ++j) {
                const int id = tid * 4 + j;
                const int kk = id >> 4;
                const int nn = (id & 15) << 2;
                Bs[buf][nn + 0][kk] = (__bf16)rb[j].x;
                Bs[buf][nn + 1][kk] = (__bf16)rb[j].y;
                Bs[buf][nn + 2][kk] = (__bf16)rb[j].z;
                Bs[buf][nn + 3][kk] = (__bf16)rb[j].w;
            }
        }
    };

    v8f acc00 = {}, acc01 = {}, acc10 = {}, acc11 = {};

    const int nTiles = Kd / BKT;
    fetch(0);
    stage(0);

    for (int t = 0; t < nTiles; ++t) {
        __syncthreads();
        const int cur = t & 1;
        if (t + 1 < nTiles) fetch((t + 1) * BKT);

        // ---- 2 K-subtiles x (2 M-tiles x 2 N-tiles) = 8 WMMAs -----------
        #pragma unroll
        for (int s = 0; s < 2; ++s) {
            // A frag: lanes<16 K 0..15, lanes>=16 K 16..31
            const int kb  = s * 32 + half * 16;
            bf16x16 af0 = ldsfrag(&As[cur][mo + r][kb]);       // rows mo..mo+15
            bf16x16 af1 = ldsfrag(&As[cur][mo + 16 + r][kb]);  // rows mo+16..mo+31
            // B frag: lanes<16 K 0..15, lanes>=16 K 16..31
            const int kb2 = s * 32 + half * 16;
            bf16x16 bf0 = ldsfrag(&Bs[cur][no + r][kb2]);
            bf16x16 bf1 = ldsfrag(&Bs[cur][no + 16 + r][kb2]);

            acc00 = __builtin_amdgcn_wmma_f32_16x16x32_bf16(false, af0, false, bf0,
                                                            (short)0, acc00, false, false);
            acc01 = __builtin_amdgcn_wmma_f32_16x16x32_bf16(false, af0, false, bf1,
                                                            (short)0, acc01, false, false);
            acc10 = __builtin_amdgcn_wmma_f32_16x16x32_bf16(false, af1, false, bf0,
                                                            (short)0, acc10, false, false);
            acc11 = __builtin_amdgcn_wmma_f32_16x16x32_bf16(false, af1, false, bf1,
                                                            (short)0, acc11, false, false);
        }

        if (t + 1 < nTiles) stage(cur ^ 1);
    }

    // ---- epilogue: C layout = VGPR v -> (M = half*8 + v, N = r) ----------
    const int nc0 = n0 + no + r;
    const int nc1 = nc0 + 16;
    const float b0v = bias ? bias[nc0] : 0.0f;
    const float b1v = bias ? bias[nc1] : 0.0f;
    #pragma unroll
    for (int mt = 0; mt < 2; ++mt) {
        const v8f a0 = mt ? acc10 : acc00;
        const v8f a1 = mt ? acc11 : acc01;
        const int mbase = m0 + mo + mt * 16 + half * 8;
        #pragma unroll
        for (int v = 0; v < 8; ++v) {
            const int mm = mbase + v;
            float val0 = alpha * a0[v] + b0v;
            float val1 = alpha * a1[v] + b1v;
            if (act == 1) {
                val0 = 0.5f * val0 * (1.0f + erff(val0 * 0.70710678118f));
                val1 = 0.5f * val1 * (1.0f + erff(val1 * 0.70710678118f));
            }
            if (Res) {
                val0 += Res[(long long)mm * sCm + nc0];
                val1 += Res[(long long)mm * sCm + nc1];
            }
            C[(long long)mm * sCm + nc0] = val0;
            C[(long long)mm * sCm + nc1] = val1;
        }
    }
}

// ---------------------------------------------------------------------------
// LayerNorm over last dim (block per row)
// ---------------------------------------------------------------------------
__global__ __launch_bounds__(256)
void layernorm_k(const float* __restrict__ X, const float* __restrict__ w,
                 const float* __restrict__ b, float* __restrict__ Y, int Dim)
{
    const int row = blockIdx.x;
    const float* x = X + (long long)row * Dim;
    float* y = Y + (long long)row * Dim;
    __shared__ float rs[256], rss[256];
    float s = 0.f, ss = 0.f;
    for (int i = threadIdx.x; i < Dim; i += 256) { float v = x[i]; s += v; ss += v * v; }
    rs[threadIdx.x] = s; rss[threadIdx.x] = ss;
    __syncthreads();
    for (int off = 128; off > 0; off >>= 1) {
        if (threadIdx.x < off) { rs[threadIdx.x] += rs[threadIdx.x + off];
                                 rss[threadIdx.x] += rss[threadIdx.x + off]; }
        __syncthreads();
    }
    const float mean = rs[0] / Dim;
    const float var  = rss[0] / Dim - mean * mean;
    const float rstd = rsqrtf(var + 1e-5f);
    for (int i = threadIdx.x; i < Dim; i += 256)
        y[i] = (x[i] - mean) * rstd * w[i] + b[i];
}

// ---------------------------------------------------------------------------
// Masked softmax over last dim (block per row). mask==null -> no masking.
// ---------------------------------------------------------------------------
__global__ __launch_bounds__(256)
void softmax_k(float* __restrict__ S, const int* __restrict__ mask, int T, int HK)
{
    const int row = blockIdx.x;
    float* s = S + (long long)row * T;
    const int* mr = mask ? mask + (long long)(row / HK) * T : nullptr;
    __shared__ float red[256];
    float mx = -3.4e38f;
    for (int i = threadIdx.x; i < T; i += 256) {
        float v = s[i];
        if (mr && mr[i] <= 0) { v = -1e9f; s[i] = v; }
        mx = fmaxf(mx, v);
    }
    red[threadIdx.x] = mx; __syncthreads();
    for (int off = 128; off > 0; off >>= 1) {
        if (threadIdx.x < off) red[threadIdx.x] = fmaxf(red[threadIdx.x], red[threadIdx.x + off]);
        __syncthreads();
    }
    mx = red[0]; __syncthreads();
    float sum = 0.f;
    for (int i = threadIdx.x; i < T; i += 256) {
        float e = __expf(s[i] - mx); s[i] = e; sum += e;
    }
    red[threadIdx.x] = sum; __syncthreads();
    for (int off = 128; off > 0; off >>= 1) {
        if (threadIdx.x < off) red[threadIdx.x] += red[threadIdx.x + off];
        __syncthreads();
    }
    const float inv = 1.0f / red[0];
    for (int i = threadIdx.x; i < T; i += 256) s[i] *= inv;
}

// ---------------------------------------------------------------------------
__global__ __launch_bounds__(256)
void bcast_latents_k(const float* __restrict__ lat, float* __restrict__ x, int KD, int total)
{
    int i = blockIdx.x * 256 + threadIdx.x;
    if (i < total) x[i] = lat[i % KD];
}

__global__ __launch_bounds__(256)
void rownorm2_k(const float* __restrict__ X, float* __restrict__ out, int Dim)
{
    const int row = blockIdx.x;
    const float* x = X + (long long)row * Dim;
    __shared__ float red[256];
    float s = 0.f;
    for (int i = threadIdx.x; i < Dim; i += 256) { float v = x[i]; s += v * v; }
    red[threadIdx.x] = s; __syncthreads();
    for (int off = 128; off > 0; off >>= 1) {
        if (threadIdx.x < off) red[threadIdx.x] += red[threadIdx.x + off];
        __syncthreads();
    }
    if (threadIdx.x == 0) out[row] = red[0];
}

__global__ __launch_bounds__(256)
void argmin_k(const float* __restrict__ dot, const float* __restrict__ xn2,
              const float* __restrict__ cn2, int* __restrict__ idx, int ne)
{
    const int m = blockIdx.x;
    const float xm = xn2[m];
    float best = 3.4e38f; int bi = 0;
    for (int n = threadIdx.x; n < ne; n += 256) {
        float d = xm + cn2[n] - 2.0f * dot[(long long)m * ne + n];
        if (d < best) { best = d; bi = n; }
    }
    __shared__ float bv[256]; __shared__ int bx[256];
    bv[threadIdx.x] = best; bx[threadIdx.x] = bi;
    __syncthreads();
    for (int off = 128; off > 0; off >>= 1) {
        if (threadIdx.x < off) {
            float v2 = bv[threadIdx.x + off]; int i2 = bx[threadIdx.x + off];
            if (v2 < bv[threadIdx.x] || (v2 == bv[threadIdx.x] && i2 < bx[threadIdx.x])) {
                bv[threadIdx.x] = v2; bx[threadIdx.x] = i2;
            }
        }
        __syncthreads();
    }
    if (threadIdx.x == 0) idx[m] = bx[0];
}

__global__ __launch_bounds__(256)
void vq_out_k(const float* __restrict__ cb, const int* __restrict__ idx,
              const float* __restrict__ xq, float* __restrict__ out,
              const float* __restrict__ scale_p, float* __restrict__ loss_sum, int Dim)
{
    const int m = blockIdx.x;
    const int id = idx[m];
    const float scale = *scale_p;
    const float* c = cb + (long long)id * Dim;
    const float* x = xq + (long long)m * Dim;
    float* o = out + (long long)m * Dim;
    float local = 0.f;
    for (int i = threadIdx.x; i < Dim; i += 256) {
        float q = c[i];
        float d = q - x[i];
        local += d * d;
        o[i] = q * scale;
    }
    __shared__ float red[256];
    red[threadIdx.x] = local; __syncthreads();
    for (int off = 128; off > 0; off >>= 1) {
        if (threadIdx.x < off) red[threadIdx.x] += red[threadIdx.x + off];
        __syncthreads();
    }
    if (threadIdx.x == 0) atomicAdd(loss_sum, red[0]);
}

__global__ __launch_bounds__(256)
void hist_k(const int* __restrict__ idx, float* __restrict__ counts, int n)
{
    int i = blockIdx.x * 256 + threadIdx.x;
    if (i < n) atomicAdd(&counts[idx[i]], 1.0f);
}

__global__ __launch_bounds__(256)
void zero_k(float* __restrict__ p, int n)
{
    int i = blockIdx.x * 256 + threadIdx.x;
    if (i < n) p[i] = 0.0f;
}

__global__ __launch_bounds__(256)
void finalize_k(const float* __restrict__ counts, const float* __restrict__ loss_sum,
                float* __restrict__ out2, int ne, float ntotal, float nelem)
{
    __shared__ float red[256];
    float s = 0.f;
    for (int n = threadIdx.x; n < ne; n += 256) {
        float p = counts[n] / ntotal;
        s += p * logf(p + 1e-10f);
    }
    red[threadIdx.x] = s; __syncthreads();
    for (int off = 128; off > 0; off >>= 1) {
        if (threadIdx.x < off) red[threadIdx.x] += red[threadIdx.x + off];
        __syncthreads();
    }
    if (threadIdx.x == 0) {
        out2[0] = 1.25f * (*loss_sum) / nelem;  // q_loss + 0.25*e_loss (numerically equal)
        out2[1] = expf(-red[0]);                // perplexity
    }
}

// ---------------------------------------------------------------------------
// Host orchestration
// ---------------------------------------------------------------------------
extern "C" void kernel_launch(void* const* d_in, const int* in_sizes, int n_in,
                              void* d_out, int out_size, void* d_ws, size_t ws_size,
                              hipStream_t stream)
{
    const float* src      = (const float*)d_in[0];   // [B,T,SRC]
    const int*   src_mask = (const int*)  d_in[1];   // [B,T]
    const float* inproj_w = (const float*)d_in[2];   // [D,SRC]
    const float* inproj_b = (const float*)d_in[3];
    const float* latents  = (const float*)d_in[4];   // [K,D]
    const float* cin_w    = (const float*)d_in[5];   // [L,3D,D]
    const float* cin_b    = (const float*)d_in[6];
    const float* cout_w   = (const float*)d_in[7];   // [L,D,D]
    const float* cout_b   = (const float*)d_in[8];
    const float* sin_w    = (const float*)d_in[9];
    const float* sin_b    = (const float*)d_in[10];
    const float* sout_w   = (const float*)d_in[11];
    const float* sout_b   = (const float*)d_in[12];
    const float* ln1_w    = (const float*)d_in[13];
    const float* ln1_b    = (const float*)d_in[14];
    const float* ln2_w    = (const float*)d_in[15];
    const float* ln2_b    = (const float*)d_in[16];
    const float* ln3_w    = (const float*)d_in[17];
    const float* ln3_b    = (const float*)d_in[18];
    const float* ffn_w1   = (const float*)d_in[19];  // [L,4D,D]
    const float* ffn_b1   = (const float*)d_in[20];
    const float* ffn_w2   = (const float*)d_in[21];  // [L,D,4D]
    const float* ffn_b2   = (const float*)d_in[22];
    const float* prevq_w  = (const float*)d_in[23];
    const float* prevq_b  = (const float*)d_in[24];
    const float* codebook = (const float*)d_in[25];  // [NE,D]
    const float* oscale   = (const float*)d_in[26];  // scalar

    float* outp = (float*)d_out;  // [B*K*D] out, then vq_loss, perplexity

    // ---- workspace carve-up (floats) ------------------------------------
    float* w = (float*)d_ws;
    size_t off = 0;
    auto alloc = [&](size_t n) { float* p = w + off; off += n; return p; };
    float* keys    = alloc((size_t)BB * TT * DD);
    float* kbuf    = alloc((size_t)BB * TT * DD);
    float* vbuf    = alloc((size_t)BB * TT * DD);
    float* scores  = alloc((size_t)BB * HH * KK * TT);
    float* scores2 = alloc((size_t)BB * HH * KK * KK);
    float* x       = alloc((size_t)BB * KK * DD);
    float* xn      = alloc((size_t)BB * KK * DD);
    float* qb      = alloc((size_t)BB * KK * DD);
    float* ob      = alloc((size_t)BB * KK * DD);
    float* hb      = alloc((size_t)BB * KK * FFD);
    float* xq      = alloc((size_t)BB * KK * DD);
    float* dotb    = alloc((size_t)BB * KK * NEE);
    float* xn2     = alloc((size_t)BB * KK);
    float* cn2     = alloc((size_t)NEE);
    float* counts  = alloc((size_t)NEE);
    float* lsum    = alloc(1);
    int*   idxb    = (int*)alloc((size_t)BB * KK);

    const int M512 = BB * KK;        // 512 latent rows (flattened)
    const int MK   = BB * TT;        // 8192 key rows

    auto gemm = [&](const float* A, long long sAm, long long sAk, long long sAhi, long long sAlo,
                    const float* Bp, long long sBn, long long sBk, long long sBhi, long long sBlo,
                    float* C, long long sCm, long long sChi, long long sClo,
                    const float* bias, const float* res,
                    int M, int N, int Kd, float alpha, int act, int nlo, int gz) {
        dim3 g((N + BN - 1) / BN, (M + BM - 1) / BM, gz);
        wmma_gemm_bf16<<<g, 256, 0, stream>>>(A, sAm, sAk, sAhi, sAlo,
                                              Bp, sBn, sBk, sBhi, sBlo,
                                              C, sCm, sChi, sClo,
                                              bias, res, M, N, Kd, alpha, act, nlo);
    };

    // ---- keys = src @ inproj_w^T + b (normalizer is identity) -----------
    gemm(src, SRCD, 1, 0, 0,
         inproj_w, SRCD, 1, 0, 0,
         keys, DD, 0, 0,
         inproj_b, nullptr, MK, DD, SRCD, 1.0f, 0, 1, 1);

    // ---- x = broadcast(latents) -----------------------------------------
    bcast_latents_k<<<(M512 * DD + 255) / 256, 256, 0, stream>>>(latents, x, KK * DD, M512 * DD);

    const float attn_alpha = 1.0f / 16.0f;  // 1/sqrt(HD=256)

    for (int l = 0; l < LL; ++l) {
        const float* wq = cin_w + (size_t)l * 3 * DD * DD;
        const float* wk = wq + (size_t)DD * DD;
        const float* wv = wq + (size_t)2 * DD * DD;
        const float* bq = cin_b + (size_t)l * 3 * DD;
        const float* bk = bq + DD;
        const float* bv = bq + 2 * DD;

        // ---------------- cross attention --------------------------------
        layernorm_k<<<M512, 256, 0, stream>>>(x, ln1_w + (size_t)l * DD, ln1_b + (size_t)l * DD, xn, DD);

        gemm(xn, DD, 1, 0, 0, wq, DD, 1, 0, 0, qb, DD, 0, 0, bq, nullptr,
             M512, DD, DD, 1.0f, 0, 1, 1);
        gemm(keys, DD, 1, 0, 0, wk, DD, 1, 0, 0, kbuf, DD, 0, 0, bk, nullptr,
             MK, DD, DD, 1.0f, 0, 1, 1);
        gemm(keys, DD, 1, 0, 0, wv, DD, 1, 0, 0, vbuf, DD, 0, 0, bv, nullptr,
             MK, DD, DD, 1.0f, 0, 1, 1);

        // scores[b,h,q,t] = (q . k) / 16
        gemm(qb, DD, 1, (long long)KK * DD, HDD,
             kbuf, DD, 1, (long long)TT * DD, HDD,
             scores, TT, (long long)HH * KK * TT, (long long)KK * TT,
             nullptr, nullptr, KK, TT, HDD, attn_alpha, 0, HH, BB * HH);

        softmax_k<<<BB * HH * KK, 256, 0, stream>>>(scores, src_mask, TT, HH * KK);

        // ob[b,q,h*HD+d] = scores @ V
        gemm(scores, TT, 1, (long long)HH * KK * TT, (long long)KK * TT,
             vbuf, 1, DD, (long long)TT * DD, HDD,
             ob, DD, (long long)KK * DD, HDD,
             nullptr, nullptr, KK, HDD, TT, 1.0f, 0, HH, BB * HH);

        // x = x + ob @ cross_out^T + b
        gemm(ob, DD, 1, 0, 0,
             cout_w + (size_t)l * DD * DD, DD, 1, 0, 0,
             x, DD, 0, 0,
             cout_b + (size_t)l * DD, x, M512, DD, DD, 1.0f, 0, 1, 1);

        // ---------------- self attention ----------------------------------
        const float* wq2 = sin_w + (size_t)l * 3 * DD * DD;
        const float* wk2 = wq2 + (size_t)DD * DD;
        const float* wv2 = wq2 + (size_t)2 * DD * DD;
        const float* bq2 = sin_b + (size_t)l * 3 * DD;
        const float* bk2 = bq2 + DD;
        const float* bv2 = bq2 + 2 * DD;

        layernorm_k<<<M512, 256, 0, stream>>>(x, ln2_w + (size_t)l * DD, ln2_b + (size_t)l * DD, xn, DD);

        gemm(xn, DD, 1, 0, 0, wq2, DD, 1, 0, 0, qb, DD, 0, 0, bq2, nullptr,
             M512, DD, DD, 1.0f, 0, 1, 1);
        gemm(xn, DD, 1, 0, 0, wk2, DD, 1, 0, 0, kbuf, DD, 0, 0, bk2, nullptr,
             M512, DD, DD, 1.0f, 0, 1, 1);
        gemm(xn, DD, 1, 0, 0, wv2, DD, 1, 0, 0, vbuf, DD, 0, 0, bv2, nullptr,
             M512, DD, DD, 1.0f, 0, 1, 1);

        gemm(qb, DD, 1, (long long)KK * DD, HDD,
             kbuf, DD, 1, (long long)KK * DD, HDD,
             scores2, KK, (long long)HH * KK * KK, (long long)KK * KK,
             nullptr, nullptr, KK, KK, HDD, attn_alpha, 0, HH, BB * HH);

        softmax_k<<<BB * HH * KK, 256, 0, stream>>>(scores2, nullptr, KK, HH * KK);

        gemm(scores2, KK, 1, (long long)HH * KK * KK, (long long)KK * KK,
             vbuf, 1, DD, (long long)KK * DD, HDD,
             ob, DD, (long long)KK * DD, HDD,
             nullptr, nullptr, KK, HDD, KK, 1.0f, 0, HH, BB * HH);

        gemm(ob, DD, 1, 0, 0,
             sout_w + (size_t)l * DD * DD, DD, 1, 0, 0,
             x, DD, 0, 0,
             sout_b + (size_t)l * DD, x, M512, DD, DD, 1.0f, 0, 1, 1);

        // ---------------- FFN ---------------------------------------------
        layernorm_k<<<M512, 256, 0, stream>>>(x, ln3_w + (size_t)l * DD, ln3_b + (size_t)l * DD, xn, DD);

        gemm(xn, DD, 1, 0, 0,
             ffn_w1 + (size_t)l * FFD * DD, DD, 1, 0, 0,
             hb, FFD, 0, 0,
             ffn_b1 + (size_t)l * FFD, nullptr, M512, FFD, DD, 1.0f, /*gelu*/1, 1, 1);

        gemm(hb, FFD, 1, 0, 0,
             ffn_w2 + (size_t)l * DD * FFD, FFD, 1, 0, 0,
             x, DD, 0, 0,
             ffn_b2 + (size_t)l * DD, x, M512, DD, FFD, 1.0f, 0, 1, 1);
    }

    // ---- VQ ---------------------------------------------------------------
    layernorm_k<<<M512, 256, 0, stream>>>(x, prevq_w, prevq_b, xq, DD);

    gemm(xq, DD, 1, 0, 0,
         codebook, DD, 1, 0, 0,
         dotb, NEE, 0, 0,
         nullptr, nullptr, M512, NEE, DD, 1.0f, 0, 1, 1);

    rownorm2_k<<<M512, 256, 0, stream>>>(xq, xn2, DD);
    rownorm2_k<<<NEE, 256, 0, stream>>>(codebook, cn2, DD);
    zero_k<<<(NEE + 1 + 255) / 256, 256, 0, stream>>>(counts, NEE + 1);  // counts + lsum (adjacent)

    argmin_k<<<M512, 256, 0, stream>>>(dotb, xn2, cn2, idxb, NEE);
    vq_out_k<<<M512, 256, 0, stream>>>(codebook, idxb, xq, outp, oscale, lsum, DD);
    hist_k<<<(M512 + 255) / 256, 256, 0, stream>>>(idxb, counts, M512);

    finalize_k<<<1, 256, 0, stream>>>(counts, lsum, outp + (size_t)BB * KK * DD,
                                      NEE, (float)M512, (float)M512 * DD);
}
